// CFnetFilter_38826504356689
// MI455X (gfx1250) — matrix operand order repaired and checked by reading, not compiled, not run on hardware
//
#include <hip/hip_runtime.h>
#include <hip/hip_bf16.h>

typedef float  v8f   __attribute__((ext_vector_type(8)));
typedef float  v2fv  __attribute__((ext_vector_type(2)));
typedef __bf16 v2bf  __attribute__((ext_vector_type(2)));
typedef __bf16 v8bf  __attribute__((ext_vector_type(8)));
typedef __bf16 v16bf __attribute__((ext_vector_type(16)));

#define TM    128      // rows per workgroup tile
#define KD    128      // inner dim
#define NF    128      // filters
#define SA    140      // fp32 LDS stride for A/h tile (12-pattern -> conflict-free frags)
#define SWT   152      // bf16 LDS stride for transposed weights (conflict-free b128 frags)
#define LOG2C 0.6931471805599453f

// split a pair of fp32 into packed bf16 hi dword + bf16 lo dword (RNE, hw cvt_pk path)
__device__ __forceinline__ void split2(float x0, float x1, unsigned& h, unsigned& l) {
    v2fv xp = { x0, x1 };
    v2bf hp = __builtin_convertvector(xp, v2bf);       // v_cvt_pk_bf16_f32
    union { v2bf b; unsigned u; } hu; hu.b = hp;
    union { unsigned u; float f; } e0, e1;
    e0.u = hu.u << 16;                                 // (float)hi0, exact
    e1.u = hu.u & 0xFFFF0000u;                         // (float)hi1, exact
    v2fv lp = { x0 - e0.f, x1 - e1.f };
    v2bf lq = __builtin_convertvector(lp, v2bf);
    union { v2bf b; unsigned u; } lu; lu.b = lq;
    h = hu.u; l = lu.u;
}

__device__ __forceinline__ float sspf(float x) {
    // shifted softplus: softplus(x) - log(2), numerically stable
    float e = __expf(-fabsf(x));
    return fmaxf(x, 0.0f) + __logf(1.0f + e) - LOG2C;
}

// B fragment: 16 consecutive bf16 (column of W^T), two aligned 16B chunks
__device__ __forceinline__ v16bf ldbfrag(const unsigned short* p) {
    v8bf a = *(const v8bf*)(p);
    v8bf b = *(const v8bf*)(p + 8);
    v16bf r;
    #pragma unroll
    for (int i = 0; i < 8; ++i) { r[i] = a[i]; r[i + 8] = b[i]; }
    return r;
}

union frag16 { v16bf bf; unsigned u[8]; };

template<bool SECOND>
__device__ __forceinline__ void gemm_pass(float* lA,
        const unsigned short* __restrict__ Wh,
        const unsigned short* __restrict__ Wl,
        const float* __restrict__ bias,
        const int* sv, float* __restrict__ out,
        int mrow, int lm, int half)
{
    // ---- build hi/lo bf16 A-fragments from this wave's private fp32 strip ----
    // A frags use contiguous k order chosen here (matches B): lanes<16: K {0..15}; lanes>=16: K {16..31}
    frag16 ahi[4], alo[4];
    const float* rowp = &lA[(mrow + lm) * SA];
    #pragma unroll
    for (int ks = 0; ks < 4; ++ks) {
        const int base = ks * 32 + half * 16;
        float4 c0 = *(const float4*)&rowp[base];
        float4 c1 = *(const float4*)&rowp[base + 4];
        float4 c2 = *(const float4*)&rowp[base + 8];
        float4 c3 = *(const float4*)&rowp[base + 12];
        // pairs (2p,2p+1) are consecutive K -> exactly one packed dword each
        split2(c0.x, c0.y, ahi[ks].u[0], alo[ks].u[0]);
        split2(c0.z, c0.w, ahi[ks].u[1], alo[ks].u[1]);
        split2(c1.x, c1.y, ahi[ks].u[2], alo[ks].u[2]);
        split2(c1.z, c1.w, ahi[ks].u[3], alo[ks].u[3]);
        split2(c2.x, c2.y, ahi[ks].u[4], alo[ks].u[4]);
        split2(c2.z, c2.w, ahi[ks].u[5], alo[ks].u[5]);
        split2(c3.x, c3.y, ahi[ks].u[6], alo[ks].u[6]);
        split2(c3.z, c3.w, ahi[ks].u[7], alo[ks].u[7]);
    }

    #pragma unroll 1
    for (int nt = 0; nt < 8; ++nt) {
        const int n0 = nt * 16;
        // B frag: column n0+lm of W^T; lanes>=16 take the K+16 half
        const unsigned short* ch = &Wh[(n0 + lm) * SWT + half * 16];
        const unsigned short* cl = &Wl[(n0 + lm) * SWT + half * 16];
        v8f ahh = {}, ahl = {}, alh = {};      // 3 independent accumulate chains
        #pragma unroll
        for (int ks = 0; ks < 4; ++ks) {
            v16bf bh = ldbfrag(ch + ks * 32);
            v16bf bl = ldbfrag(cl + ks * 32);
            ahh = __builtin_amdgcn_wmma_f32_16x16x32_bf16(false, ahi[ks].bf, false, bh,
                                                          (short)0, ahh, false, false);
            ahl = __builtin_amdgcn_wmma_f32_16x16x32_bf16(false, ahi[ks].bf, false, bl,
                                                          (short)0, ahl, false, false);
            alh = __builtin_amdgcn_wmma_f32_16x16x32_bf16(false, alo[ks].bf, false, bh,
                                                          (short)0, alh, false, false);
        }
        const float bv = bias[n0 + lm];
        if constexpr (!SECOND) {
            // h = ssp(.) written back fp32 into the wave-private strip (conflict-free)
            #pragma unroll
            for (int v = 0; v < 8; ++v) {
                float xx = ahh[v] + (ahl[v] + alh[v]) + bv;
                lA[(mrow + half * 8 + v) * SA + n0 + lm] = sspf(xx);
            }
        } else {
            // sorted-segment running sum down the 8 rows this lane-half owns
            float run = 0.0f;
            #pragma unroll
            for (int v = 0; v < 8; ++v) {
                run += sspf(ahh[v] + (ahl[v] + alh[v]) + bv);
                bool flush = (v == 7) || (sv[v + 1] != sv[v]);
                if (flush) {
                    if (sv[v] >= 0)
                        atomicAdd(&out[sv[v] * NF + n0 + lm], run);
                    run = 0.0f;
                }
            }
        }
    }
}

extern "C" __global__ __launch_bounds__(256)
void cfnet_fused_bf16x3_kernel(const float* __restrict__ dijk,
                               const int*   __restrict__ segj,
                               const float* __restrict__ W1,
                               const float* __restrict__ b1,
                               const float* __restrict__ W2,
                               const float* __restrict__ b2,
                               float*       __restrict__ out,
                               int nrows)
{
    __shared__ __align__(16) float          lA  [TM * SA];    //  71,680 B
    __shared__ __align__(16) unsigned short lW1h[NF * SWT];   //  38,912 B each
    __shared__ __align__(16) unsigned short lW1l[NF * SWT];
    __shared__ __align__(16) unsigned short lW2h[NF * SWT];
    __shared__ __align__(16) unsigned short lW2l[NF * SWT];
    __shared__ int lseg[TM];                                  // total 227,840 B

    const int tid     = threadIdx.x;
    const int rowBase = blockIdx.x * TM;

    // ---- stage A tile (fp32, coalesced float4) ----
    {
        const float4* gA = (const float4*)dijk;
        #pragma unroll
        for (int j = 0; j < (TM * (KD / 4)) / 256; ++j) {
            int i = tid + j * 256;
            int r = i >> 5, c4 = i & 31;
            int gr = rowBase + r;
            float4 v = make_float4(0.f, 0.f, 0.f, 0.f);
            if (gr < nrows) v = gA[gr * (KD / 4) + c4];
            *(float4*)&lA[r * SA + c4 * 4] = v;
        }
    }
    // ---- stage W1/W2: transpose + hi/lo bf16 split, dword-packed stores ----
    #pragma unroll 1
    for (int w = 0; w < 2; ++w) {
        const float* gW = w ? W2 : W1;
        unsigned short* th = w ? lW2h : lW1h;
        unsigned short* tl = w ? lW2l : lW1l;
        #pragma unroll 4
        for (int j = 0; j < (NF * (KD / 2)) / 256; ++j) {     // 32 iters
            int i  = tid + j * 256;
            int kp = i >> 7;          // k-pair 0..63 (fixed per wave -> coalesced)
            int n  = i & 127;
            float x0 = gW[(2 * kp) * NF + n];
            float x1 = gW[(2 * kp + 1) * NF + n];
            unsigned hv, lv;
            split2(x0, x1, hv, lv);
            *(unsigned*)&th[n * SWT + 2 * kp] = hv;
            *(unsigned*)&tl[n * SWT + 2 * kp] = lv;
        }
    }
    if (tid < TM) {
        int gr = rowBase + tid;
        lseg[tid] = (gr < nrows) ? segj[gr] : -1;
    }
    __syncthreads();
    // past this point everything is wave-private except read-only lW*/lseg

    const int lane = tid & 31;
    const int wave = tid >> 5;
    const int half = lane >> 4;
    const int lm   = lane & 15;
    const int mrow = wave * 16;

    int sv[8];
    const int rbase = mrow + half * 8;
    #pragma unroll
    for (int v = 0; v < 8; ++v) sv[v] = lseg[rbase + v];

    gemm_pass<false>(lA, lW1h, lW1l, b1, sv, out, mrow, lm, half);
    gemm_pass<true >(lA, lW2h, lW2l, b2, sv, out, mrow, lm, half);
}

extern "C" void kernel_launch(void* const* d_in, const int* in_sizes, int n_in,
                              void* d_out, int out_size, void* d_ws, size_t ws_size,
                              hipStream_t stream) {
    const float* dijk = (const float*)d_in[0];
    const int*   segj = (const int*)  d_in[1];
    const float* W1   = (const float*)d_in[2];
    const float* b1   = (const float*)d_in[3];
    const float* W2   = (const float*)d_in[4];
    const float* b2   = (const float*)d_in[5];
    float* out = (float*)d_out;

    const int nrows = in_sizes[0] / KD;   // 2,000,000

    // output accumulated via atomics -> zero it every call (graph-capturable memset)
    hipMemsetAsync(out, 0, (size_t)out_size * sizeof(float), stream);

    const int nblocks = (nrows + TM - 1) / TM;  // 15625
    hipLaunchKernelGGL(cfnet_fused_bf16x3_kernel, dim3(nblocks), dim3(256), 0, stream,
                       dijk, segj, W1, b1, W2, b2, out, nrows);
}
